// AttentionExpert_12893491823135
// MI455X (gfx1250) — matrix operand order repaired and checked by reading, not compiled, not run on hardware
//
#include <hip/hip_runtime.h>
#include <cstdint>

// Problem constants (match reference)
#define BB   4
#define SS   2048
#define HH   1024
#define NHD  16
#define HDD  64
#define INV_SCALE 0.125f   // 1/sqrt(64)

typedef __attribute__((ext_vector_type(16))) __bf16 bf16x16;
typedef __attribute__((ext_vector_type(8)))  float  f32x8;

#define WMMA_BF16(a, b, c) \
  __builtin_amdgcn_wmma_f32_16x16x32_bf16(false, (a), false, (b), (short)0, (c), false, false)

// CDNA5 async global->LDS path (ASYNCcnt-tracked), guarded so the build can
// never regress if the toolchain lacks the builtins.
#if defined(__has_builtin)
#  if __has_builtin(__builtin_amdgcn_global_load_async_to_lds_b128) && \
      __has_builtin(__builtin_amdgcn_s_wait_asynccnt)
#    define USE_ASYNC_LDS 1
#  endif
#endif
#ifndef USE_ASYNC_LDS
#  define USE_ASYNC_LDS 0
#endif

#if USE_ASYNC_LDS
// Builtin signature (from hipcc diagnostic): (int4 AS1*, int4 AS3*, Imm, Imm)
typedef int vint4 __attribute__((vector_size(16)));
typedef __attribute__((address_space(1))) vint4 GInt4;
typedef __attribute__((address_space(3))) vint4 LInt4;
#endif

union FragU { bf16x16 v; uint4 q[2]; };

// A-matrix 16x32 bf16 fragment (ISA 7.12.2 dense 16-bit A layout):
// lane = 16*hf + r holds row r; elems e0..7 = K[hf*8 .. hf*8+7], e8..15 = K[16+hf*8 ..]
__device__ inline bf16x16 load_fragA(const __bf16* rowbase, int k0, int hf) {
  FragU f;
  f.q[0] = *reinterpret_cast<const uint4*>(rowbase + k0 + hf * 8);
  f.q[1] = *reinterpret_cast<const uint4*>(rowbase + k0 + 16 + hf * 8);
  return f.v;
}
// B-matrix 32x16 bf16 fragment: lane holds column n=r of B (i.e. row n of B^T),
// elems e0..7 = K[hf*8 .. hf*8+7], e8..15 = K[16+hf*8 ..] (same as A).
__device__ inline bf16x16 load_fragB(const __bf16* rowbase, int k0, int hf) {
  FragU f;
  f.q[0] = *reinterpret_cast<const uint4*>(rowbase + k0 + hf * 8);
  f.q[1] = *reinterpret_cast<const uint4*>(rowbase + k0 + 16 + hf * 8);
  return f.v;
}

// ---------------------------------------------------------------------------
// fp32 -> bf16 convert (fully coalesced)
__global__ __launch_bounds__(256)
void k_cvt_bf16(const float* __restrict__ in, __bf16* __restrict__ out, int n) {
  int i = blockIdx.x * 256 + threadIdx.x;
  if (i < n) out[i] = (__bf16)in[i];
}

// W[1024][1024] fp32 -> Wt[n][k] bf16, LDS-tiled 32x32 transpose so both the
// global reads and the global writes are coalesced (naive version would
// amplify HBM traffic ~32x through 4KB-strided reads).
__global__ __launch_bounds__(256)
void k_transpose_cvt(const float* __restrict__ W, __bf16* __restrict__ Wt) {
  __shared__ float tile[32][33];           // +1 col: no LDS bank conflicts
  const int tx  = threadIdx.x & 31;
  const int ty4 = threadIdx.x >> 5;        // 8 row-groups of 4
  const int k0 = blockIdx.x * 32;          // source row block (k)
  const int n0 = blockIdx.y * 32;          // source col block (n)
#pragma unroll
  for (int i = 0; i < 4; ++i) {
    const int row = ty4 * 4 + i;
    tile[row][tx] = W[(size_t)(k0 + row) * HH + n0 + tx];
  }
  __syncthreads();
#pragma unroll
  for (int i = 0; i < 4; ++i) {
    const int row = ty4 * 4 + i;           // n offset within tile
    Wt[(size_t)(n0 + row) * HH + k0 + tx] = (__bf16)tile[tx][row];
  }
}

// ---------------------------------------------------------------------------
// GEMM: C[m][n] = A[m][:] . Wt[n][:] + bias[n]
// block = 256 thr = 8 waves; wave tile 32(M) x 64(N); block tile 128x128.
// QKV variant stores bf16 head-major: out[((b*NH+h)*S+s)*HD + d]
__global__ __launch_bounds__(256)
void k_gemm_qkv(const __bf16* __restrict__ A, const __bf16* __restrict__ Wt,
                const float* __restrict__ bias, __bf16* __restrict__ outHM) {
  const int lane = threadIdx.x & 31;
  const int w    = threadIdx.x >> 5;
  const int r    = lane & 15, hf = lane >> 4;
  const int rowA = blockIdx.x * 128 + (w & 3) * 32;
  const int rowB = blockIdx.y * 128 + (w >> 2) * 64;
  const f32x8 z = {0.f, 0.f, 0.f, 0.f, 0.f, 0.f, 0.f, 0.f};
  f32x8 acc[2][4];
#pragma unroll
  for (int i = 0; i < 2; ++i)
#pragma unroll
    for (int j = 0; j < 4; ++j) acc[i][j] = z;

  const __bf16* a0p = A  + (size_t)(rowA + r)      * HH;
  const __bf16* a1p = A  + (size_t)(rowA + 16 + r) * HH;
  const __bf16* b0p = Wt + (size_t)(rowB + r)      * HH;
  const __bf16* b1p = Wt + (size_t)(rowB + 16 + r) * HH;
  const __bf16* b2p = Wt + (size_t)(rowB + 32 + r) * HH;
  const __bf16* b3p = Wt + (size_t)(rowB + 48 + r) * HH;

  for (int k0 = 0; k0 < HH; k0 += 32) {
    bf16x16 a0 = load_fragA(a0p, k0, hf);
    bf16x16 a1 = load_fragA(a1p, k0, hf);
    bf16x16 b0 = load_fragB(b0p, k0, hf);
    bf16x16 b1 = load_fragB(b1p, k0, hf);
    bf16x16 b2 = load_fragB(b2p, k0, hf);
    bf16x16 b3 = load_fragB(b3p, k0, hf);
    acc[0][0] = WMMA_BF16(a0, b0, acc[0][0]);
    acc[0][1] = WMMA_BF16(a0, b1, acc[0][1]);
    acc[0][2] = WMMA_BF16(a0, b2, acc[0][2]);
    acc[0][3] = WMMA_BF16(a0, b3, acc[0][3]);
    acc[1][0] = WMMA_BF16(a1, b0, acc[1][0]);
    acc[1][1] = WMMA_BF16(a1, b1, acc[1][1]);
    acc[1][2] = WMMA_BF16(a1, b2, acc[1][2]);
    acc[1][3] = WMMA_BF16(a1, b3, acc[1][3]);
  }
  // C/D layout: lane = 16*hf + r ; VGPR e -> M = hf*8+e (per 16x16 tile), N = r
#pragma unroll
  for (int i = 0; i < 2; ++i) {
#pragma unroll
    for (int j = 0; j < 4; ++j) {
      const int n  = rowB + j * 16 + r;
      const float bn = bias[n];
      const int h = n >> 6, d = n & 63;
#pragma unroll
      for (int e = 0; e < 8; ++e) {
        const int m = rowA + i * 16 + hf * 8 + e;
        const int b = m >> 11, s = m & (SS - 1);
        outHM[(((size_t)b * NHD + h) * SS + s) * HDD + d] =
            (__bf16)(acc[i][j][e] + bn);
      }
    }
  }
}

// Output-projection variant: fp32 store, token-major [m][n] = [B*S][H]
__global__ __launch_bounds__(256)
void k_gemm_out(const __bf16* __restrict__ A, const __bf16* __restrict__ Wt,
                const float* __restrict__ bias, float* __restrict__ out) {
  const int lane = threadIdx.x & 31;
  const int w    = threadIdx.x >> 5;
  const int r    = lane & 15, hf = lane >> 4;
  const int rowA = blockIdx.x * 128 + (w & 3) * 32;
  const int rowB = blockIdx.y * 128 + (w >> 2) * 64;
  const f32x8 z = {0.f, 0.f, 0.f, 0.f, 0.f, 0.f, 0.f, 0.f};
  f32x8 acc[2][4];
#pragma unroll
  for (int i = 0; i < 2; ++i)
#pragma unroll
    for (int j = 0; j < 4; ++j) acc[i][j] = z;

  const __bf16* a0p = A  + (size_t)(rowA + r)      * HH;
  const __bf16* a1p = A  + (size_t)(rowA + 16 + r) * HH;
  const __bf16* b0p = Wt + (size_t)(rowB + r)      * HH;
  const __bf16* b1p = Wt + (size_t)(rowB + 16 + r) * HH;
  const __bf16* b2p = Wt + (size_t)(rowB + 32 + r) * HH;
  const __bf16* b3p = Wt + (size_t)(rowB + 48 + r) * HH;

  for (int k0 = 0; k0 < HH; k0 += 32) {
    bf16x16 a0 = load_fragA(a0p, k0, hf);
    bf16x16 a1 = load_fragA(a1p, k0, hf);
    bf16x16 b0 = load_fragB(b0p, k0, hf);
    bf16x16 b1 = load_fragB(b1p, k0, hf);
    bf16x16 b2 = load_fragB(b2p, k0, hf);
    bf16x16 b3 = load_fragB(b3p, k0, hf);
    acc[0][0] = WMMA_BF16(a0, b0, acc[0][0]);
    acc[0][1] = WMMA_BF16(a0, b1, acc[0][1]);
    acc[0][2] = WMMA_BF16(a0, b2, acc[0][2]);
    acc[0][3] = WMMA_BF16(a0, b3, acc[0][3]);
    acc[1][0] = WMMA_BF16(a1, b0, acc[1][0]);
    acc[1][1] = WMMA_BF16(a1, b1, acc[1][1]);
    acc[1][2] = WMMA_BF16(a1, b2, acc[1][2]);
    acc[1][3] = WMMA_BF16(a1, b3, acc[1][3]);
  }
#pragma unroll
  for (int i = 0; i < 2; ++i) {
#pragma unroll
    for (int j = 0; j < 4; ++j) {
      const int n  = rowB + j * 16 + r;
      const float bn = bias[n];
#pragma unroll
      for (int e = 0; e < 8; ++e) {
        const int m = rowA + i * 16 + hf * 8 + e;
        out[(size_t)m * HH + n] = acc[i][j][e] + bn;
      }
    }
  }
}

// ---------------------------------------------------------------------------
// Banded attention: one wave per (b, h, 16-query tile). Window = 80 keys
// (5 key tiles of 16), padded to 96 for three 32-deep P.V WMMA steps.
__global__ __launch_bounds__(32)
void k_attn(const __bf16* __restrict__ Qh, const __bf16* __restrict__ Kh,
            const __bf16* __restrict__ Vh, const int* __restrict__ amask,
            __bf16* __restrict__ ctx) {
  __shared__ float  sP[16][96];   // probabilities, cols 80..95 zero-padded
  __shared__ __bf16 sV[96][64];   // V rows for window (zero if out of range)

  const int lane = threadIdx.x;
  const int r = lane & 15, hf = lane >> 4;
  const int q0  = blockIdx.x * 16;
  const int h   = blockIdx.y;
  const int b   = blockIdx.z;
  const int kt0 = q0 - 32;

  const size_t headoff = ((size_t)b * NHD + h) * SS * HDD;
  const __bf16* Q = Qh + headoff;
  const __bf16* K = Kh + headoff;
  const __bf16* V = Vh + headoff;

  // Stage 96x64 V slab into LDS (uint4 = 8 bf16 per lane per iter).
  // In-range rows go through the CDNA5 async global->LDS data mover path
  // (ASYNCcnt); out-of-range rows are zero-filled via DS stores (disjoint
  // addresses, so no ordering hazard between the two paths).
  for (int it = 0; it < 24; ++it) {
    int i = it * 32 + lane;
    int row = i >> 3, c8 = (i & 7) * 8;
    int key = kt0 + row;
    if (key >= 0 && key < SS) {
#if USE_ASYNC_LDS
      __builtin_amdgcn_global_load_async_to_lds_b128(
          (GInt4*)(V + (size_t)key * HDD + c8),
          (LInt4*)&sV[row][c8], 0, 0);
#else
      *reinterpret_cast<uint4*>(&sV[row][c8]) =
          *reinterpret_cast<const uint4*>(V + (size_t)key * HDD + c8);
#endif
    } else {
      uint4 zz = {0u, 0u, 0u, 0u};
      *reinterpret_cast<uint4*>(&sV[row][c8]) = zz;
    }
  }

  // Q fragments (16 queries x 64 dims, two 32-deep chunks)
  const __bf16* qrow = Q + (size_t)(q0 + r) * HDD;
  bf16x16 aq0 = load_fragA(qrow, 0, hf);
  bf16x16 aq1 = load_fragA(qrow, 32, hf);

  // Scores: 5 key tiles x 2 WMMAs, then mask
  const f32x8 z = {0.f, 0.f, 0.f, 0.f, 0.f, 0.f, 0.f, 0.f};
  float sc[5][8];
#pragma unroll
  for (int t = 0; t < 5; ++t) {
    const int key = kt0 + t * 16 + r;          // this lane's key (B-frag column)
    const int kcl = min(max(key, 0), SS - 1);
    const __bf16* krow = K + (size_t)kcl * HDD;
    bf16x16 bk0 = load_fragB(krow, 0, hf);
    bf16x16 bk1 = load_fragB(krow, 32, hf);
    f32x8 acc = z;
    acc = WMMA_BF16(aq0, bk0, acc);
    acc = WMMA_BF16(aq1, bk1, acc);
    const int am = (key >= 0 && key < SS) ? amask[b * SS + key] : 0;
#pragma unroll
    for (int e = 0; e < 8; ++e) {
      const int qi = q0 + hf * 8 + e;
      const int di = qi - key;
      const bool valid = (key >= 0) && (key < SS) && (di <= 32) && (di >= -32) && (am != 0);
      sc[t][e] = valid ? acc[e] * INV_SCALE : -3.4e38f;
    }
  }

  // Row softmax: row of C tile lives across the 16 lanes of this half-wave
#pragma unroll
  for (int e = 0; e < 8; ++e) {
    float mx = -3.4e38f;
#pragma unroll
    for (int t = 0; t < 5; ++t) mx = fmaxf(mx, sc[t][e]);
    mx = fmaxf(mx, __shfl_xor(mx, 1, 32));
    mx = fmaxf(mx, __shfl_xor(mx, 2, 32));
    mx = fmaxf(mx, __shfl_xor(mx, 4, 32));
    mx = fmaxf(mx, __shfl_xor(mx, 8, 32));
    float sum = 0.f;
    float ex[5];
#pragma unroll
    for (int t = 0; t < 5; ++t) {
      ex[t] = (sc[t][e] > -1.0e37f) ? __expf(sc[t][e] - mx) : 0.f;
      sum += ex[t];
    }
    sum += __shfl_xor(sum, 1, 32);
    sum += __shfl_xor(sum, 2, 32);
    sum += __shfl_xor(sum, 4, 32);
    sum += __shfl_xor(sum, 8, 32);
    const float inv = (sum > 0.f) ? (1.0f / sum) : 0.f;  // nan_to_num rows -> 0
    const int row = hf * 8 + e;
#pragma unroll
    for (int t = 0; t < 5; ++t) sP[row][t * 16 + r] = ex[t] * inv;
    sP[row][80 + r] = 0.f;   // zero pad cols 80..95
  }
#if USE_ASYNC_LDS
  __builtin_amdgcn_s_wait_asynccnt(0);   // async V slab landed in LDS
#endif
  __syncthreads();

  // out = P (16x96) . V (96x64): 3 K-chunks x 4 N-tiles of WMMA
  f32x8 oacc[4] = {z, z, z, z};
#pragma unroll
  for (int kk = 0; kk < 3; ++kk) {
    FragU fa;
#pragma unroll
    for (int e = 0; e < 8; ++e) {
      fa.v[e]     = (__bf16)sP[r][kk * 32 + hf * 8 + e];
      fa.v[e + 8] = (__bf16)sP[r][kk * 32 + 16 + hf * 8 + e];
    }
#pragma unroll
    for (int nt = 0; nt < 4; ++nt) {
      const int d = nt * 16 + r;
      bf16x16 bv;
#pragma unroll
      for (int e = 0; e < 16; ++e) bv[e] = sV[kk * 32 + hf * 8 + e + (e >= 8 ? 8 : 0)][d];
      oacc[nt] = WMMA_BF16(fa.v, bv, oacc[nt]);
    }
  }
  // store context token-major [b*S+q][h*64+d] (bf16)
#pragma unroll
  for (int nt = 0; nt < 4; ++nt) {
#pragma unroll
    for (int e = 0; e < 8; ++e) {
      const int qrowi = q0 + hf * 8 + e;
      const int d = nt * 16 + r;
      ctx[((size_t)(b * SS + qrowi)) * HH + h * HDD + d] = (__bf16)oacc[nt][e];
    }
  }
}

// ---------------------------------------------------------------------------
extern "C" void kernel_launch(void* const* d_in, const int* in_sizes, int n_in,
                              void* d_out, int out_size, void* d_ws, size_t ws_size,
                              hipStream_t stream) {
  (void)in_sizes; (void)n_in; (void)out_size; (void)ws_size;
  const float* x    = (const float*)d_in[0];
  const int*   am   = (const int*)  d_in[1];
  const float* Wq   = (const float*)d_in[2];
  const float* bq   = (const float*)d_in[3];
  const float* Wk   = (const float*)d_in[4];
  const float* bk   = (const float*)d_in[5];
  const float* Wv   = (const float*)d_in[6];
  const float* bv   = (const float*)d_in[7];
  const float* Wo   = (const float*)d_in[8];
  const float* bo   = (const float*)d_in[9];
  float* out = (float*)d_out;

  char* ws = (char*)d_ws;
  const size_t MB = 1u << 20;
  __bf16* xb  = (__bf16*)(ws);              // 16 MB, reused as ctx after QKV
  __bf16* qh  = (__bf16*)(ws + 16 * MB);    // 16 MB
  __bf16* kh  = (__bf16*)(ws + 32 * MB);    // 16 MB
  __bf16* vh  = (__bf16*)(ws + 48 * MB);    // 16 MB
  __bf16* wt  = (__bf16*)(ws + 64 * MB);    //  2 MB (reused per weight, stream-ordered)
  __bf16* ctx = xb;

  const int NTOK = BB * SS * HH;            // 8M elements
  k_cvt_bf16<<<NTOK / 256, 256, 0, stream>>>(x, xb, NTOK);

  dim3 gG(BB * SS / 128, HH / 128), bG(256);
  dim3 gT(HH / 32, HH / 32);

  k_transpose_cvt<<<gT, 256, 0, stream>>>(Wq, wt);
  k_gemm_qkv<<<gG, bG, 0, stream>>>(xb, wt, bq, qh);
  k_transpose_cvt<<<gT, 256, 0, stream>>>(Wk, wt);
  k_gemm_qkv<<<gG, bG, 0, stream>>>(xb, wt, bk, kh);
  k_transpose_cvt<<<gT, 256, 0, stream>>>(Wv, wt);
  k_gemm_qkv<<<gG, bG, 0, stream>>>(xb, wt, bv, vh);

  dim3 gA(SS / 16, NHD, BB);
  k_attn<<<gA, 32, 0, stream>>>(qh, kh, vh, am, ctx);

  k_transpose_cvt<<<gT, 256, 0, stream>>>(Wo, wt);
  k_gemm_out<<<gG, bG, 0, stream>>>(ctx, wt, bo, out);
}
